// RefLocal6_59150289600836
// MI455X (gfx1250) — matrix operand order repaired and checked by reading, not compiled, not run on hardware
//
#include <hip/hip_runtime.h>

// ---------------------------------------------------------------------------
// CDNA5 (gfx1250) fused local-attention, bf16 WMMA (16x16x32), wave32.
// ---------------------------------------------------------------------------

typedef __bf16 bf16;
typedef bf16  bf16x8  __attribute__((ext_vector_type(8)));
typedef bf16  bf16x16 __attribute__((ext_vector_type(16)));
typedef float f32x8   __attribute__((ext_vector_type(8)));

union BF16Frag {
    bf16x16 v;
    bf16x8  h[2];
};

// Convert 8 consecutive f32 -> bf16 elements [dst, dst+8) of a fragment.
__device__ inline void cvt8(const float* __restrict__ s, BF16Frag& f, int dst) {
    float4 a = *(const float4*)(s);
    float4 b = *(const float4*)(s + 4);
    f.v[dst + 0] = (bf16)a.x; f.v[dst + 1] = (bf16)a.y;
    f.v[dst + 2] = (bf16)a.z; f.v[dst + 3] = (bf16)a.w;
    f.v[dst + 4] = (bf16)b.x; f.v[dst + 5] = (bf16)b.y;
    f.v[dst + 6] = (bf16)b.z; f.v[dst + 7] = (bf16)b.w;
}

// ---------------------------------------------------------------------------
// Kernel 1: Y[M,128] (bf16) = X[M,128] (f32) * W[128,128] (f32)
// 8 waves / block, one 16-row M-tile per wave. W^T staged in LDS as bf16.
// ---------------------------------------------------------------------------
__global__ __launch_bounds__(256)
void conv1x1_bf16(const float* __restrict__ X,
                  const float* __restrict__ W,
                  bf16* __restrict__ Y,
                  int M)
{
    __shared__ __align__(16) bf16 Wt[128 * 128];   // Wt[n*128 + k] = W[k][n]

    const int tid = threadIdx.x;
    for (int e = tid; e < 128 * 128; e += 256) {
        int k = e >> 7, n = e & 127;
        Wt[n * 128 + k] = (bf16)W[e];
    }
    __syncthreads();

    const int lane = tid & 31;
    const int wave = tid >> 5;
    const int g    = lane >> 4;     // half-wave select
    const int ln   = lane & 15;
    const int row0 = blockIdx.x * 128 + wave * 16;
    if (row0 >= M) return;

    // A fragments (16x32 bf16 per K-chunk): lane half g picks K runs
    // {g*8..g*8+7} and {16+g*8..16+g*8+7} within each chunk of 32.
    BF16Frag a[4];
    const float* xrow = X + (size_t)(row0 + ln) * 128;
    for (int kc = 0; kc < 4; ++kc) {
        cvt8(xrow + kc * 32 + g * 8,      a[kc], 0);
        cvt8(xrow + kc * 32 + 16 + g * 8, a[kc], 8);
    }

    f32x8 acc[8];
    for (int nt = 0; nt < 8; ++nt) acc[nt] = {0.f,0.f,0.f,0.f,0.f,0.f,0.f,0.f};

    for (int kc = 0; kc < 4; ++kc) {
        for (int nt = 0; nt < 8; ++nt) {
            // B fragment (32x16): lane col n = nt*16+ln, K runs kc*32 + g*8 .. +7 and kc*32 + 16 + g*8 .. +7
            BF16Frag bfr;
            const bf16* bp = &Wt[(nt * 16 + ln) * 128 + kc * 32 + g * 8];
            bfr.h[0] = *(const bf16x8*)(bp);
            bfr.h[1] = *(const bf16x8*)(bp + 16);
            acc[nt] = __builtin_amdgcn_wmma_f32_16x16x32_bf16(
                false, a[kc].v, false, bfr.v, (short)0, acc[nt], false, false);
        }
    }

    // C/D layout: VGPR v holds row v + 8*g, col = nt*16 + ln
    for (int nt = 0; nt < 8; ++nt)
        for (int v = 0; v < 8; ++v)
            Y[(size_t)(row0 + v + g * 8) * 128 + nt * 16 + ln] = (bf16)acc[nt][v];
}

// ---------------------------------------------------------------------------
// Kernel 2: fused 5x5 local attention. One wave per 16-pixel tile (same row).
// The tile's 25-key windows span 5x20 = 100 keys -> padded to 112 (7 N-tiles).
//   S(16x112) = Q(16x128) * K^T          (28 WMMAs)
//   per-pixel softmax over its banded 25 entries -> dense P(16x128) bf16
//   O(16x64)  = P(16x128) * V(128x64)    (16 WMMAs)
// ---------------------------------------------------------------------------
__global__ __launch_bounds__(64)
void local_attn(const bf16* __restrict__ Q,    // [B*H*W, 128]
                const bf16* __restrict__ Kf,   // [B*H*W, 128]
                const float* __restrict__ V,   // [B*H*W, 64]
                float* __restrict__ O,         // [B*H*W, 64]
                int Bn, int Hn, int Wn)
{
    __shared__ __align__(16) float Ssc[2][16 * 112];
    __shared__ __align__(16) bf16  Pm [2][16 * 128];
    __shared__ __align__(16) bf16  Vt [2][64 * 128];   // Vt[bin*128 + key]

    const int tid  = threadIdx.x;
    const int lane = tid & 31;
    const int wave = tid >> 5;
    const int g    = lane >> 4;
    const int ln   = lane & 15;

    const int NT   = Wn / 16;
    const int tile = blockIdx.x * 2 + wave;
    const int wt = tile % NT;
    const int h  = (tile / NT) % Hn;
    const int b  = tile / (NT * Hn);
    const int w0 = wt * 16;
    const size_t plane = (size_t)(b * Hn + h) * Wn;

    float* Sw = Ssc[wave];
    bf16*  Pw = Pm[wave];
    bf16*  Vw = Vt[wave];

    // ---- Q fragments (kept for all 7 score N-tiles) ----
    BF16Frag qa[4];
    const bf16* qrow = Q + (plane + w0 + ln) * 128;
    for (int kc = 0; kc < 4; ++kc) {
        qa[kc].h[0] = *(const bf16x8*)(qrow + kc * 32 + g * 8);
        qa[kc].h[1] = *(const bf16x8*)(qrow + kc * 32 + 16 + g * 8);
    }

    // ---- scores: S = Q * K^T over 112 padded key slots ----
    for (int nt = 0; nt < 7; ++nt) {
        const int nk = nt * 16 + ln;           // this lane's B-matrix column (key id)
        const int di = nk / 20, t = nk % 20;
        const int hh = h + di - 2, ww = w0 + t - 2;
        const bool ok = (nk < 100) && (hh >= 0) && (hh < Hn) && (ww >= 0) && (ww < Wn);
        const bf16* krow = Kf + ((size_t)(b * Hn + hh) * Wn + ww) * 128;

        f32x8 acc = {0.f,0.f,0.f,0.f,0.f,0.f,0.f,0.f};
        for (int kc = 0; kc < 4; ++kc) {
            BF16Frag bfr;
            if (ok) {
                bfr.h[0] = *(const bf16x8*)(krow + kc * 32 + g * 8);
                bfr.h[1] = *(const bf16x8*)(krow + kc * 32 + 16 + g * 8);
            } else {
                for (int i = 0; i < 16; ++i) bfr.v[i] = (bf16)0.f;
            }
            acc = __builtin_amdgcn_wmma_f32_16x16x32_bf16(
                false, qa[kc].v, false, bfr.v, (short)0, acc, false, false);
        }
        for (int v = 0; v < 8; ++v)
            Sw[(v + g * 8) * 112 + nt * 16 + ln] = acc[v];
    }

    // ---- stage V^T into LDS, zero-padded to 128 key slots (real zeros:
    //      0 * garbage could be NaN, so padding must be written) ----
    for (int e = lane; e < 64 * 128; e += 32) {
        const int key = e >> 6;        // e = key*64 + bin  -> coalesced global reads
        const int bin = e & 63;
        float val = 0.f;
        if (key < 100) {
            const int di = key / 20, t = key % 20;
            const int hh = h + di - 2, ww = w0 + t - 2;
            if (hh >= 0 && hh < Hn && ww >= 0 && ww < Wn)
                val = V[((size_t)(b * Hn + hh) * Wn + ww) * 64 + bin];
        }
        Vw[bin * 128 + key] = (bf16)val;
    }

    // ---- softmax per pixel (lanes 0..15), build dense P row ----
    if (lane < 16) {
        const int p = lane;
        float sc[25];
        float mx = -3.4e38f;
        for (int di = 0; di < 5; ++di)
            for (int dj = 0; dj < 5; ++dj) {
                float s = Sw[p * 112 + di * 20 + p + dj];
                sc[di * 5 + dj] = s;
                mx = fmaxf(mx, s);
            }
        float sum = 0.f;
        for (int i = 0; i < 25; ++i) { sc[i] = __expf(sc[i] - mx); sum += sc[i]; }
        const float inv = 1.f / sum;

        bf16x8 z8;
        for (int i = 0; i < 8; ++i) z8[i] = (bf16)0.f;
        for (int c = 0; c < 128; c += 8) *(bf16x8*)&Pw[p * 128 + c] = z8;
        for (int di = 0; di < 5; ++di)
            for (int dj = 0; dj < 5; ++dj)
                Pw[p * 128 + di * 20 + p + dj] = (bf16)(sc[di * 5 + dj] * inv);
    }
    // Same-wave DS ops are in-order: no barrier needed before reading Pw/Vw.

    // ---- O = P * V ----
    f32x8 oacc[4];
    for (int nt = 0; nt < 4; ++nt) oacc[nt] = {0.f,0.f,0.f,0.f,0.f,0.f,0.f,0.f};
    for (int kc = 0; kc < 4; ++kc) {
        BF16Frag pa;
        const bf16* prow = &Pw[ln * 128];
        pa.h[0] = *(const bf16x8*)(prow + kc * 32 + g * 8);
        pa.h[1] = *(const bf16x8*)(prow + kc * 32 + 16 + g * 8);
        for (int nt = 0; nt < 4; ++nt) {
            BF16Frag vb;
            const bf16* vrow = &Vw[(nt * 16 + ln) * 128 + kc * 32 + g * 8];
            vb.h[0] = *(const bf16x8*)(vrow);
            vb.h[1] = *(const bf16x8*)(vrow + 16);
            oacc[nt] = __builtin_amdgcn_wmma_f32_16x16x32_bf16(
                false, pa.v, false, vb.v, (short)0, oacc[nt], false, false);
        }
    }
    for (int nt = 0; nt < 4; ++nt)
        for (int v = 0; v < 8; ++v)
            O[(plane + w0 + v + g * 8) * 64 + nt * 16 + ln] = oacc[nt][v];
}

// ---------------------------------------------------------------------------
extern "C" void kernel_launch(void* const* d_in, const int* in_sizes, int n_in,
                              void* d_out, int out_size, void* d_ws, size_t ws_size,
                              hipStream_t stream) {
    (void)in_sizes; (void)n_in; (void)out_size; (void)ws_size;

    const float* mainp = (const float*)d_in[0];
    const float* refp  = (const float*)d_in[1];
    const float* refv  = (const float*)d_in[2];
    const float* Wm    = (const float*)d_in[3];
    const float* Wr    = (const float*)d_in[4];
    float* out = (float*)d_out;

    const int B = 4, H = 128, W = 128;
    const int M = B * H * W;                 // 65536 pixels

    bf16* qm = (bf16*)d_ws;                  // [M,128] bf16 : conv_main
    bf16* kr = qm + (size_t)M * 128;         // [M,128] bf16 : conv_ref  (32 MB total)

    conv1x1_bf16<<<M / 128, 256, 0, stream>>>(mainp, Wm, qm, M);
    conv1x1_bf16<<<M / 128, 256, 0, stream>>>(refp,  Wr, kr, M);

    const int tiles = B * H * (W / 16);      // 4096 tiles, 2 waves per block
    local_attn<<<tiles / 2, 64, 0, stream>>>(qm, kr, refv, out, B, H, W);
}
